// Network_25692494365407
// MI455X (gfx1250) — matrix operand order repaired and checked by reading, not compiled, not run on hardware
//
#include <hip/hip_runtime.h>
#include <hip/hip_bf16.h>

// ---------------- problem constants (from reference) ----------------
#define Bsz   16
#define Tt    100
#define Nn    10240
#define NA    8192
#define KB_AB (Nn / 32)   // 320 k-blocks of 32 for Wab
#define KB_ST (NA / 32)   // 256 k-blocks for W_stp
#define JT_AB (Nn / 16)   // 640 column tiles
#define JT_ST (NA / 16)   // 512 column tiles with STP contribution

#define DT_C        0.01f
#define USE_C       0.03f
#define INV_TAU_FAC 1.0f      // 1/TAU_FAC, TAU_FAC=1.0
#define INV_TAU_REC 4.0f      // 1/TAU_REC, TAU_REC=0.25
#define E_SYN       0.8f      // EXP_DT_TAU_SYN
#define E_TAU       0.9f      // EXP_DT_TAU
#define E_AD        0.99f     // EXP_ADAPT
#define A_AD        0.1f      // A_ADAPT

typedef __bf16 bf16_t;
typedef __attribute__((ext_vector_type(16))) __bf16 v16bf;
typedef __attribute__((ext_vector_type(8)))  __bf16 v8bf;
typedef __attribute__((ext_vector_type(8)))  float  v8f;
typedef __attribute__((ext_vector_type(4)))  unsigned int u32x4;

struct B32raw { u32x4 lo, hi; };   // 32 bytes == v16bf

// A operand, 16-bit A 16x32 layout: lane holds row m=lane&15,
// elements 0..7 = K at (half*8 + 0..7), elements 8..15 = K at (half*8+16..23).
__device__ __forceinline__ v16bf load_a16(const bf16_t* p) {
    v8bf lo = *(const v8bf*)p;
    v8bf hi = *(const v8bf*)(p + 16);
    v16bf a;
#pragma unroll
    for (int e = 0; e < 8; ++e) { a[e] = lo[e]; a[e + 8] = hi[e]; }
    return a;
}

// B operand from pre-packed tiles: 32 contiguous bytes per lane.
__device__ __forceinline__ v16bf load_b_nt(const bf16_t* p) {  // streaming (Wab)
    B32raw raw;
    raw.lo = __builtin_nontemporal_load((const u32x4*)p);
    raw.hi = __builtin_nontemporal_load((const u32x4*)p + 1);
    return __builtin_bit_cast(v16bf, raw);
}
__device__ __forceinline__ v16bf load_b_rt(const bf16_t* p) {  // L2-resident (W_stp)
    return *(const v16bf*)p;
}

__device__ __forceinline__ v8f wmma_bf16(v16bf a, v16bf b, v8f c) {
    return __builtin_amdgcn_wmma_f32_16x16x32_bf16(
        false, a, false, b, (short)0, c, false, false);
}

// -------------------------------------------------------------------
// Pack fp32 weights (K x rowStride row-major) into bf16 WMMA-B tiles:
// out[((j*KB+kb)*32+lane)*16+e] = W[(kb*32+(lane>>4)*8+e+(e>=8?8:0))*rowStride + j*16+(lane&15)]
// Dense 16-bit B 32x16 layout: lanes 0-15 hold K=0..7,16..23, lanes 16-31 K=8..15,24..31.
// -------------------------------------------------------------------
__global__ __launch_bounds__(256) void pack_weights_kernel(
    const float* __restrict__ W, bf16_t* __restrict__ out,
    int KB, int rowStride, int tilesJ)
{
    int g = blockIdx.x * 256 + threadIdx.x;
    int total = tilesJ * KB * 32;
    if (g >= total) return;
    int lane = g & 31;
    int kb   = (g >> 5) % KB;
    int j    = (g >> 5) / KB;
    int col  = j * 16 + (lane & 15);
    int krow = kb * 32 + (lane >> 4) * 8;
    size_t base = (size_t)g * 16;
#pragma unroll
    for (int e = 0; e < 16; ++e)
        out[base + e] = (bf16_t)W[(size_t)(krow + e + (e >= 8 ? 8 : 0)) * rowStride + col];
}

// ---------------- one-time init: rates0 + first STP update ----------
__global__ __launch_bounds__(256) void init_state_kernel(
    const float* __restrict__ ff,
    const float* __restrict__ rec0, const float* __restrict__ th0,
    const float* __restrict__ u0,   const float* __restrict__ x0,
    float* __restrict__ rates_s, float* __restrict__ rec_s,
    float* __restrict__ th_s, float* __restrict__ u_s, float* __restrict__ x_s,
    bf16_t* __restrict__ rates_bf, bf16_t* __restrict__ aux_bf)
{
    int idx = blockIdx.x * 256 + threadIdx.x;
    if (idx >= Bsz * Nn) return;
    int b = idx / Nn, n = idx % Nn;
    float r0 = fmaxf(ff[(size_t)b * Tt * Nn + n] + rec0[idx] - th0[idx], 0.0f);
    rates_s[idx]  = r0;
    rec_s[idx]    = rec0[idx];
    th_s[idx]     = th0[idx];
    rates_bf[idx] = (bf16_t)r0;
    if (n < NA) {                       // STP update for step t=0
        int si = b * NA + n;
        float u = u0[si], x = x0[si];
        float un = u + DT_C * ((USE_C - u) * INV_TAU_FAC + USE_C * (1.0f - u) * r0);
        float xn = x + DT_C * ((1.0f - x) * INV_TAU_REC - un * x * r0);
        u_s[si] = un;
        x_s[si] = xn;
        aux_bf[si] = (bf16_t)(un * xn * r0);
    }
}

// -------------------------------------------------------------------
// Fused GEMM + recurrent epilogue + next-step STP update, one timestep.
// Block = 256 threads = 8 waves, one block per 16-column tile j.
// 2-deep software pipeline, manually unrolled x2 (two independent
// buffer pairs) so the compiler needs no rotation copies and up to
// 8 b128 loads stay outstanding per wave while WMMAs execute.
// -------------------------------------------------------------------
__global__ __launch_bounds__(256) void gemm_step_kernel(
    const bf16_t* __restrict__ wab_p,   // packed (JT_AB,KB_AB,32,16)
    const bf16_t* __restrict__ wstp_p,  // packed (JT_ST,KB_ST,32,16)
    bf16_t* __restrict__ rates_bf,      // (16,N)  A operand, step t
    bf16_t* __restrict__ aux_bf,        // (16,NA) A operand, step t
    const float*  __restrict__ ff,      // (B,T,N)
    float* __restrict__ rec_s, float* __restrict__ th_s,
    float* __restrict__ rates_s,
    float* __restrict__ u_s, float* __restrict__ x_s,
    float* __restrict__ out, int t)
{
    const int j    = blockIdx.x;
    const int tid  = threadIdx.x;
    const int wave = tid >> 5;
    const int lane = tid & 31;
    const int half = lane >> 4;
    const int nloc = lane & 15;
    const int m    = nloc;

    v8f c = {};

    // ---- rates @ Wab_T : K = 10240, split over 8 waves ----
    {
        const bf16_t* arow = rates_bf + (size_t)m * Nn + half * 8;
        const bf16_t* brow = wab_p + ((size_t)j * KB_AB * 32 + lane) * 16;
        int kb = wave;
        v16bf a0 = load_a16(arow + kb * 32);
        v16bf b0 = load_b_nt(brow + (size_t)kb * 512);
        v16bf a1 = load_a16(arow + (kb + 8) * 32);
        v16bf b1 = load_b_nt(brow + (size_t)(kb + 8) * 512);
        for (kb += 16; kb < KB_AB; kb += 16) {
            c  = wmma_bf16(a0, b0, c);
            a0 = load_a16(arow + kb * 32);
            b0 = load_b_nt(brow + (size_t)kb * 512);
            c  = wmma_bf16(a1, b1, c);
            a1 = load_a16(arow + (kb + 8) * 32);
            b1 = load_b_nt(brow + (size_t)(kb + 8) * 512);
        }
        c = wmma_bf16(a0, b0, c);
        c = wmma_bf16(a1, b1, c);
    }

    // ---- aux @ W_stp : first 512 tiles only, K = 8192 ----
    if (j < JT_ST) {
        const bf16_t* arow = aux_bf + (size_t)m * NA + half * 8;
        const bf16_t* brow = wstp_p + ((size_t)j * KB_ST * 32 + lane) * 16;
        int kb = wave;
        v16bf a0 = load_a16(arow + kb * 32);
        v16bf b0 = load_b_rt(brow + (size_t)kb * 512);
        v16bf a1 = load_a16(arow + (kb + 8) * 32);
        v16bf b1 = load_b_rt(brow + (size_t)(kb + 8) * 512);
        for (kb += 16; kb < KB_ST; kb += 16) {
            c  = wmma_bf16(a0, b0, c);
            a0 = load_a16(arow + kb * 32);
            b0 = load_b_rt(brow + (size_t)kb * 512);
            c  = wmma_bf16(a1, b1, c);
            a1 = load_a16(arow + (kb + 8) * 32);
            b1 = load_b_rt(brow + (size_t)(kb + 8) * 512);
        }
        c = wmma_bf16(a0, b0, c);
        c = wmma_bf16(a1, b1, c);
    }

    // ---- split-K reduction across the 8 waves in LDS ----
    __shared__ float red[8 * 256];
#pragma unroll
    for (int r = 0; r < 8; ++r)
        red[wave * 256 + lane * 8 + r] = c[r];
    __syncthreads();

    float h = 0.0f;
#pragma unroll
    for (int w = 0; w < 8; ++w)
        h += red[w * 256 + tid];

    // map flat C element tid -> (batch b, column col)
    const int r   = tid & 7;
    const int ln  = tid >> 3;
    const int b   = r + ((ln >> 4) << 3);
    const int col = j * 16 + (ln & 15);

    // ---- fused recurrent epilogue ----
    const size_t si = (size_t)b * Nn + col;
    const size_t fi = ((size_t)b * Tt + t) * Nn + col;
    float rec = rec_s[si];
    float th  = th_s[si];
    float rt  = rates_s[si];

    float recn = rec * E_SYN + h * (1.0f - E_SYN);
    float net  = ff[fi] + recn;
    float nl   = fmaxf(net - th, 0.0f);
    float rtn  = rt * E_TAU + nl * (1.0f - E_TAU);
    float thn  = th * E_AD + rtn * (A_AD * (1.0f - E_AD));

    rec_s[si]   = recn;
    rates_s[si] = rtn;
    th_s[si]    = thn;
    out[fi]     = rtn;

    // ---- fused next-step STP update + bf16 A operands for step t+1 ----
    rates_bf[si] = (bf16_t)rtn;
    if (col < NA) {
        int ui = b * NA + col;
        float u = u_s[ui], x = x_s[ui];
        float un = u + DT_C * ((USE_C - u) * INV_TAU_FAC + USE_C * (1.0f - u) * rtn);
        float xn = x + DT_C * ((1.0f - x) * INV_TAU_REC - un * x * rtn);
        u_s[ui] = un;
        x_s[ui] = xn;
        aux_bf[ui] = (bf16_t)(un * xn * rtn);
    }
}

// -------------------------------------------------------------------
extern "C" void kernel_launch(void* const* d_in, const int* in_sizes, int n_in,
                              void* d_out, int out_size, void* d_ws, size_t ws_size,
                              hipStream_t stream) {
    (void)in_sizes; (void)n_in; (void)out_size; (void)ws_size;

    const float* ff   = (const float*)d_in[0];   // (B,T,N)
    const float* Wab  = (const float*)d_in[1];   // (N,N)
    const float* Wstp = (const float*)d_in[2];   // (NA,NA)
    const float* rec0 = (const float*)d_in[3];   // (B,N)
    const float* th0  = (const float*)d_in[4];   // (B,N)
    const float* u0   = (const float*)d_in[5];   // (B,NA)
    const float* x0   = (const float*)d_in[6];   // (B,NA)
    float* out = (float*)d_out;

    // scratch carve-up (256B aligned)
    char* ws = (char*)d_ws;
    size_t off = 0;
    auto carve = [&](size_t bytes) {
        char* p = ws + off;
        off += (bytes + 255) & ~(size_t)255;
        return p;
    };
    bf16_t* wab_p    = (bf16_t*)carve((size_t)Nn * Nn * sizeof(bf16_t));   // 209.7 MB
    bf16_t* wstp_p   = (bf16_t*)carve((size_t)NA * NA * sizeof(bf16_t));   // 134.2 MB
    float*  rates_s  = (float*) carve((size_t)Bsz * Nn * sizeof(float));
    float*  rec_s    = (float*) carve((size_t)Bsz * Nn * sizeof(float));
    float*  th_s     = (float*) carve((size_t)Bsz * Nn * sizeof(float));
    float*  u_s      = (float*) carve((size_t)Bsz * NA * sizeof(float));
    float*  x_s      = (float*) carve((size_t)Bsz * NA * sizeof(float));
    bf16_t* rates_bf = (bf16_t*)carve((size_t)Bsz * Nn * sizeof(bf16_t));
    bf16_t* aux_bf   = (bf16_t*)carve((size_t)Bsz * NA * sizeof(bf16_t));

    // one-time: bf16 weight packing into WMMA B-operand tile layout
    {
        int groups = JT_AB * KB_AB * 32;
        pack_weights_kernel<<<(groups + 255) / 256, 256, 0, stream>>>(
            Wab, wab_p, KB_AB, Nn, JT_AB);
    }
    {
        int groups = JT_ST * KB_ST * 32;
        pack_weights_kernel<<<(groups + 255) / 256, 256, 0, stream>>>(
            Wstp, wstp_p, KB_ST, NA, JT_ST);
    }

    // one-time: initial state + STP update / A operands for step 0
    init_state_kernel<<<(Bsz * Nn + 255) / 256, 256, 0, stream>>>(
        ff, rec0, th0, u0, x0, rates_s, rec_s, th_s, u_s, x_s,
        rates_bf, aux_bf);

    // 100 recurrent steps: single fused kernel per step
    for (int t = 0; t < Tt; ++t) {
        gemm_step_kernel<<<JT_AB, 256, 0, stream>>>(
            wab_p, wstp_p, rates_bf, aux_bf, ff,
            rec_s, th_s, rates_s, u_s, x_s, out, t);
    }
}
